// TopNWordByWord_23347442221554
// MI455X (gfx1250) — compile-verified, matrix operand order checked by reading, not run on hardware
//
#include <hip/hip_runtime.h>
#include <hip/hip_bf16.h>

typedef __attribute__((ext_vector_type(16))) _Float16 v16h;
typedef __attribute__((ext_vector_type(8)))  _Float16 v8h;
typedef __attribute__((ext_vector_type(8)))  float    v8f;

#define B_    64
#define S_    128
#define A_    4
#define W_    40
#define D_    256
#define Q_    60
#define TOPN  10
#define K_TOT (TOPN * W_)          // 400 gathered story rows
#define CHUNK 48                   // K-chunk held in LDS (3 N-tiles of 16)
#define NTILES 3
#define QROWS 64                   // Q padded to 4 M-tiles
#define QA_STRIDE 264              // 256 + 8 halves: kills row bank conflicts
#define ST_STRIDE 264
#define NWAVES 4
#define EPSV 1e-6f
#define NEG_BIG (-3.0e38f)

__global__ __launch_bounds__(128, 1)
void topn_wbw_kernel(const float* __restrict__ scores,   // [B,S,A]
                     const float* __restrict__ story,    // [B,S,W,D]
                     const float* __restrict__ qa_emb,   // [B,A,Q,D]
                     const float* __restrict__ qa_w,     // [B,A,Q,1]
                     const float* __restrict__ w_story,  // [B,S,W,1]
                     const float* __restrict__ qa_enc,   // [B,A,D]
                     const float* __restrict__ alpha,    // [1]
                     const float* __restrict__ beta,     // [1]
                     float* __restrict__ out)            // [B,A]
{
    __shared__ _Float16 qa_lds[QROWS * QA_STRIDE];   // 33792 B, normalized qa rows (f16)
    __shared__ _Float16 st_lds[CHUNK * ST_STRIDE];   // 25344 B, normalized story rows (f16)
    __shared__ float    s_scores[S_];
    __shared__ float    r_val[128];
    __shared__ int      r_idx[128];
    __shared__ int      s_top[TOPN];
    __shared__ float    s_cosmax[QROWS];
    __shared__ float    s_wq[QROWS];
    __shared__ float    s_scalar[2];                 // [1] = Sq

    const int tid  = threadIdx.x;
    const int lane = tid & 31;
    const int wave = tid >> 5;
    const int bid  = blockIdx.x;
    const int b    = bid >> 2;                       // /A_
    const int a    = bid & 3;
    const int m15  = lane & 15;                      // N column / M row within tile
    const int grp  = lane >> 4;                      // WMMA half-wave group

    // ---------------- Phase 1: top-10 selection over sentence scores ----------------
    if (tid < S_) s_scores[tid] = scores[(b * S_ + tid) * A_ + a];
    __syncthreads();
    for (int it = 0; it < TOPN; ++it) {
        r_val[tid] = s_scores[tid];
        r_idx[tid] = tid;
        __syncthreads();
        for (int off = 64; off >= 1; off >>= 1) {
            if (tid < off) {
                if (r_val[tid + off] > r_val[tid]) {
                    r_val[tid] = r_val[tid + off];
                    r_idx[tid] = r_idx[tid + off];
                }
            }
            __syncthreads();
        }
        if (tid == 0) { s_top[it] = r_idx[0]; s_scores[r_idx[0]] = NEG_BIG; }
        __syncthreads();
    }

    // ---------------- Phase 2: qa rows -> L2-normalized f16 in LDS ----------------
    const float* qa_base = qa_emb + (size_t)(b * A_ + a) * Q_ * D_;
    for (int q = wave; q < QROWS; q += NWAVES) {
        if (q < Q_) {
            const float* rowp = qa_base + (size_t)q * D_;
            float v[8]; float ss = 0.f;
            #pragma unroll
            for (int j = 0; j < 8; ++j) { v[j] = rowp[lane + 32 * j]; ss += v[j] * v[j]; }
            #pragma unroll
            for (int m = 1; m < 32; m <<= 1) ss += __shfl_xor(ss, m, 32);
            const float inv = rsqrtf(ss + EPSV);
            #pragma unroll
            for (int j = 0; j < 8; ++j)
                qa_lds[q * QA_STRIDE + lane + 32 * j] = (_Float16)(v[j] * inv);
        } else {
            #pragma unroll
            for (int j = 0; j < 8; ++j)
                qa_lds[q * QA_STRIDE + lane + 32 * j] = (_Float16)0.f;
        }
    }

    if (tid < QROWS)
        s_wq[tid] = (tid < Q_) ? qa_w[(b * A_ + a) * Q_ + tid] : 0.f;

    if (wave == 0) {  // Sq = sum(qa_encoding^2)
        const float* qe = qa_enc + (size_t)(b * A_ + a) * D_;
        float ss = 0.f;
        #pragma unroll
        for (int j = 0; j < 8; ++j) { const float x = qe[lane + 32 * j]; ss += x * x; }
        #pragma unroll
        for (int m = 1; m < 32; m <<= 1) ss += __shfl_xor(ss, m, 32);
        if (lane == 0) s_scalar[1] = ss;
    }
    __syncthreads();

    // ---------------- Main loop: gather+normalize K-chunks, WMMA, running max ----------------
    float acc[8];                     // per-lane top_enc partial (d = lane + 32j)
    float vmax[8];                    // per-lane running cos max, rows M = r + 8*grp
    #pragma unroll
    for (int j = 0; j < 8; ++j) { acc[j] = 0.f; vmax[j] = NEG_BIG; }

    const v8f czero = {};
    const int arow = wave * 16 + m15; // this wave's qa row for the A operand

    const int nchunks = (K_TOT + CHUNK - 1) / CHUNK;  // 9
    for (int c = 0; c < nchunks; ++c) {
        const int base = c * CHUNK;
        const int rows = min(CHUNK, K_TOT - base);

        // gather + L2 normalize story rows into st_lds; fold top_enc accumulation in
        for (int r = wave; r < rows; r += NWAVES) {
            const int k    = base + r;
            const int sent = s_top[k / W_];
            const int word = k % W_;
            const float* rowp = story + (((size_t)b * S_ + sent) * W_ + word) * D_;
            const float  wr   = w_story[((size_t)b * S_ + sent) * W_ + word];
            float v[8]; float ss = 0.f;
            #pragma unroll
            for (int j = 0; j < 8; ++j) { v[j] = rowp[lane + 32 * j]; ss += v[j] * v[j]; }
            #pragma unroll
            for (int m = 1; m < 32; m <<= 1) ss += __shfl_xor(ss, m, 32);
            const float inv = rsqrtf(ss + EPSV);
            #pragma unroll
            for (int j = 0; j < 8; ++j) {
                st_lds[r * ST_STRIDE + lane + 32 * j] = (_Float16)(v[j] * inv);
                acc[j] += v[j] * wr;                   // raw (un-normalized) * weight
            }
        }
        __syncthreads();

        // WMMA: one 16-row M-tile per wave, 3 N-tiles, D=256 contracted in steps of 32
        v8f c0 = czero, c1 = czero, c2 = czero;
        #pragma unroll
        for (int dstep = 0; dstep < D_ / 32; ++dstep) {
            const int d0 = dstep * 32;
            // A operand: lanes 0-15 -> K d0+0..7 & d0+16..23 ; lanes 16-31 -> +8 shift
            const v8h alo = *(const v8h*)&qa_lds[arow * QA_STRIDE + d0 + grp * 8];
            const v8h ahi = *(const v8h*)&qa_lds[arow * QA_STRIDE + d0 + 16 + grp * 8];
            const v16h av = __builtin_shufflevector(alo, ahi, 0,1,2,3,4,5,6,7,8,9,10,11,12,13,14,15);
            #pragma unroll
            for (int nt = 0; nt < NTILES; ++nt) {
                const int brow = nt * 16 + m15;        // B column n = st row (Bᵀ is row-major here)
                const v8h blo = *(const v8h*)&st_lds[brow * ST_STRIDE + d0 + grp * 8];
                const v8h bhi = *(const v8h*)&st_lds[brow * ST_STRIDE + d0 + 16 + grp * 8];
                const v16h bv = __builtin_shufflevector(blo, bhi, 0,1,2,3,4,5,6,7,8,9,10,11,12,13,14,15);
                if (nt == 0)
                    c0 = __builtin_amdgcn_wmma_f32_16x16x32_f16(false, av, false, bv, (short)0, c0, false, false);
                else if (nt == 1)
                    c1 = __builtin_amdgcn_wmma_f32_16x16x32_f16(false, av, false, bv, (short)0, c1, false, false);
                else
                    c2 = __builtin_amdgcn_wmma_f32_16x16x32_f16(false, av, false, bv, (short)0, c2, false, false);
            }
        }

        // running max with column-validity masking (stale/partial tiles masked out)
        #pragma unroll
        for (int nt = 0; nt < NTILES; ++nt) {
            const int  col   = base + nt * 16 + m15;
            const bool valid = (col < K_TOT);
            const v8f  cv    = (nt == 0) ? c0 : (nt == 1) ? c1 : c2;
            #pragma unroll
            for (int r2 = 0; r2 < 8; ++r2)
                if (valid) vmax[r2] = fmaxf(vmax[r2], cv[r2]);
        }
        __syncthreads();   // st_lds about to be overwritten by next chunk
    }

    // ---------------- Max reduce over N (within each 16-lane group) ----------------
    #pragma unroll
    for (int r2 = 0; r2 < 8; ++r2) {
        float v = vmax[r2];
        #pragma unroll
        for (int m = 1; m < 16; m <<= 1) v = fmaxf(v, __shfl_xor(v, m, 32));
        if (m15 == 0) s_cosmax[wave * 16 + grp * 8 + r2] = v;   // M = r2 + 8*grp
    }
    __syncthreads();

    // ---------------- top_enc reduce (reuse st_lds as f32 scratch) ----------------
    float* wsum = (float*)st_lds;                    // 4 waves x 256 floats = 4 KB
    #pragma unroll
    for (int j = 0; j < 8; ++j) wsum[wave * D_ + lane + 32 * j] = acc[j];
    __syncthreads();
    {
        const float t0 = wsum[tid]       + wsum[D_ + tid]       + wsum[2 * D_ + tid]       + wsum[3 * D_ + tid];
        const float t1 = wsum[tid + 128] + wsum[D_ + tid + 128] + wsum[2 * D_ + tid + 128] + wsum[3 * D_ + tid + 128];
        r_val[tid] = t0 * t0 + t1 * t1;
    }
    __syncthreads();

    // ---------------- Final combine ----------------
    if (tid == 0) {
        float St = 0.f;
        for (int i = 0; i < 128; ++i) St += r_val[i];
        const float Sq      = s_scalar[1];
        const float sum_cos = beta[0] * (Sq / (Sq + EPSV)) / sqrtf(St + EPSV);

        float wsq = 0.f;
        for (int q = 0; q < Q_; ++q) wsq += s_wq[q];
        const float invw = 1.f / (wsq + EPSV);

        float tot = 0.f;
        for (int q = 0; q < Q_; ++q) tot += s_cosmax[q] * s_wq[q] * invw;

        out[bid] = alpha[0] * tot + sum_cos;
    }
}

extern "C" void kernel_launch(void* const* d_in, const int* in_sizes, int n_in,
                              void* d_out, int out_size, void* d_ws, size_t ws_size,
                              hipStream_t stream) {
    (void)in_sizes; (void)n_in; (void)out_size; (void)d_ws; (void)ws_size;
    const float* scores  = (const float*)d_in[0];
    const float* story   = (const float*)d_in[1];
    const float* qa_emb  = (const float*)d_in[2];
    const float* qa_w    = (const float*)d_in[3];
    const float* w_story = (const float*)d_in[4];
    const float* qa_enc  = (const float*)d_in[5];
    const float* alpha   = (const float*)d_in[6];
    const float* beta    = (const float*)d_in[7];
    // d_in[8] = top_n (fixed at 10 to match LDS sizing)
    float* out = (float*)d_out;

    topn_wbw_kernel<<<B_ * A_, 128, 0, stream>>>(scores, story, qa_emb, qa_w,
                                                 w_story, qa_enc, alpha, beta, out);
}
